// CodeBook_17300128268647
// MI455X (gfx1250) — matrix operand order repaired and checked by reading, not compiled, not run on hardware
//
#include <hip/hip_runtime.h>
#include <hip/hip_bf16.h>
#include <math.h>

// Problem constants: H=8, B=4, N=2048, D=512, C=2048
#define Hh 8
#define Bb 4
#define Nn 2048
#define Dd 512
#define Cc 2048
#define BN (Bb * Nn)            // 8192 rows per head

typedef __attribute__((ext_vector_type(16))) __bf16   v16bf;
typedef __attribute__((ext_vector_type(8)))  float    v8f;
typedef __attribute__((ext_vector_type(4)))  unsigned v4u;
typedef __attribute__((ext_vector_type(4)))  int      v4i;

union Frag { v4u q[2]; v16bf v; };

// ---- workspace layout ----
// floats: x2 [0,65536) | y2 [65536,81920) | pmax [81920,+4194304) | pidx same count
// then bf16 planes for embed hi/lo (each H*C*D = 8388608 bf16)
#define X2_OFF 0
#define Y2_OFF (Hh * BN)                          // 65536
#define PM_OFF (Y2_OFF + Hh * Cc)                 // 81920
#define NPART  64                                 // 2048/32 partials per row
#define PI_OFF (PM_OFF + (size_t)Hh * BN * NPART)
#define EH_OFF (PI_OFF + (size_t)Hh * BN * NPART) // float offset where bf16 planes start
#define EPLANE ((size_t)Hh * Cc * Dd)             // elements per embed plane

// ---- d_out layout (floats): quantize | embed_ind | dist ----
#define Q_OFF    ((size_t)0)
#define IND_OFF  ((size_t)Hh * BN * Dd)           // 33554432
#define DIST_OFF (IND_OFF + (size_t)Hh * BN)      // 33619968

// ---- CDNA5 async-copy path (guarded; falls back to ds_store staging) ----
#if defined(__has_builtin)
#if __has_builtin(__builtin_amdgcn_global_load_async_to_lds_b128) && \
    __has_builtin(__builtin_amdgcn_s_wait_asynccnt)
#define USE_ASYNC 1
#endif
#endif
#ifndef USE_ASYNC
#define USE_ASYNC 0
#endif

__device__ __forceinline__ unsigned pack2_bf16(float a, float b) {
    union { __bf16 h[2]; unsigned u; } p;
    p.h[0] = (__bf16)a; p.h[1] = (__bf16)b;
    return p.u;
}
__device__ __forceinline__ float bt(float f) {   // truncate to bf16 grid (exact hi part)
    return __uint_as_float(__float_as_uint(f) & 0xFFFF0000u);
}

// ---------------------------------------------------------------------------
// Kernel 1: x row norms; embed row norms + bf16 hi/lo plane split.
// One wave per 512-float row.
// ---------------------------------------------------------------------------
__global__ __launch_bounds__(256) void prep_kernel(
    const float* __restrict__ x, const float* __restrict__ e,
    float* __restrict__ ws, unsigned short* __restrict__ eh_plane,
    unsigned short* __restrict__ el_plane)
{
    int wave = threadIdx.x >> 5;
    int lane = threadIdx.x & 31;
    int row  = blockIdx.x * 8 + wave;
    float s = 0.f;
    if (row < Hh * BN) {                         // x: norms only
        const float4* p4 = (const float4*)(x + (size_t)row * Dd);
        #pragma unroll
        for (int k = 0; k < 4; ++k) {
            float4 v = p4[lane + k * 32];
            s += v.x * v.x + v.y * v.y + v.z * v.z + v.w * v.w;
        }
        #pragma unroll
        for (int off = 16; off; off >>= 1) s += __shfl_xor(s, off, 32);
        if (lane == 0) ws[X2_OFF + row] = s;
    } else {                                     // embed: norms + split
        int rr = row - Hh * BN;
        if (rr >= Hh * Cc) return;
        const float4* p4 = (const float4*)(e + (size_t)rr * Dd);
        uint2* hq = (uint2*)(eh_plane + (size_t)rr * Dd);
        uint2* lq = (uint2*)(el_plane + (size_t)rr * Dd);
        #pragma unroll
        for (int k = 0; k < 4; ++k) {
            int q = lane + k * 32;
            float4 v = p4[q];
            s += v.x * v.x + v.y * v.y + v.z * v.z + v.w * v.w;
            float h0 = bt(v.x), h1 = bt(v.y), h2 = bt(v.z), h3 = bt(v.w);
            uint2 hu, lu;
            hu.x = pack2_bf16(h0, h1);          hu.y = pack2_bf16(h2, h3);
            lu.x = pack2_bf16(v.x-h0, v.y-h1);  lu.y = pack2_bf16(v.z-h2, v.w-h3);
            hq[q] = hu;  lq[q] = lu;
        }
        #pragma unroll
        for (int off = 16; off; off >>= 1) s += __shfl_xor(s, off, 32);
        if (lane == 0) ws[Y2_OFF + rr] = s;
    }
}

// ---------------------------------------------------------------------------
// Kernel 2: bf16x3 WMMA GEMM + fused -sqrt(x2+y2-2ab) + partial argmax.
// WG = 256 threads = 8 waves (4 M x 2 N); WG tile 128x64; wave tile 32x32.
// Double-buffered LDS; B tiles async-copied from pre-split planes; A tiles
// register-prefetched one K-step ahead and converted once per block.
// ---------------------------------------------------------------------------
#define TM   128
#define TN   64
#define TK   32
#define SA   40                       // LDS row stride in bf16 (80 B, conflict-free)
#define ABUF (TM * SA)                // 5120 shorts per A plane
#define BBUF (TN * SA)                // 2560 shorts per B plane
#define BUFS (2*ABUF + 2*BBUF)        // 15360 shorts per buffer

__global__ __launch_bounds__(256) void gemm_dist_kernel(
    const float* __restrict__ x,
    const unsigned short* __restrict__ eh_plane,
    const unsigned short* __restrict__ el_plane,
    const float* __restrict__ x2, const float* __restrict__ y2,
    float* __restrict__ dist_out,
    float* __restrict__ pmax, int* __restrict__ pidx)
{
    __shared__ __align__(16) unsigned short smem[2 * BUFS];

    const int h     = blockIdx.z;
    const int mbase = blockIdx.x * TM;
    const int nbase = blockIdx.y * TN;
    const int tid   = threadIdx.x;
    const int lane  = tid & 31;
    const int lh    = lane & 15;
    const int sel   = lane >> 4;
    const int wave  = tid >> 5;
    const int m0    = (wave & 3) * 32;
    const int n0    = (wave >> 2) * 32;

    const float* xg = x + ((size_t)h * BN + mbase) * Dd;

    // --- helpers as lambdas ---
    auto loadA = [&](float4* ar, int kt) {
        #pragma unroll
        for (int i = 0; i < 4; ++i) {
            int f4 = tid + i * 256;
            int r = f4 >> 3, c4 = (f4 & 7) * 4;
            ar[i] = *(const float4*)(xg + (size_t)r * Dd + kt * TK + c4);
        }
    };
    auto convstoreA = [&](const float4* ar, int buf) {
        unsigned short* Ah = smem + buf * BUFS;
        unsigned short* Al = Ah + ABUF;
        #pragma unroll
        for (int i = 0; i < 4; ++i) {
            int f4 = tid + i * 256;
            int r = f4 >> 3, c4 = (f4 & 7) * 4;
            float4 v = ar[i];
            float h0 = bt(v.x), h1 = bt(v.y), h2 = bt(v.z), h3 = bt(v.w);
            uint2 hu, lu;
            hu.x = pack2_bf16(h0, h1);           hu.y = pack2_bf16(h2, h3);
            lu.x = pack2_bf16(v.x-h0, v.y-h1);   lu.y = pack2_bf16(v.z-h2, v.w-h3);
            *(uint2*)(Ah + r * SA + c4) = hu;
            *(uint2*)(Al + r * SA + c4) = lu;
        }
    };
    auto stageB = [&](int kt, int buf) {
        unsigned short* Bh = smem + buf * BUFS + 2 * ABUF;
        #pragma unroll
        for (int i = 0; i < 2; ++i) {
            int id    = tid + i * 256;           // 0..511
            int plane = id >> 8;                 // 0 = hi, 1 = lo
            int rem   = id & 255;
            int r     = rem >> 2, c = (rem & 3) * 8;
            const unsigned short* g =
                (plane ? el_plane : eh_plane) +
                ((size_t)h * Cc + nbase + r) * Dd + kt * TK + c;
            unsigned short* l = Bh + plane * BBUF + r * SA + c;
#if USE_ASYNC
            __builtin_amdgcn_global_load_async_to_lds_b128(
                (__attribute__((address_space(1))) v4i*)g,
                (__attribute__((address_space(3))) v4i*)l, 0, 0);
#else
            *(v4u*)l = *(const v4u*)g;
#endif
        }
    };

    v8f acc[2][2] = {};
    float4 ar[4];

    // prologue: stage tile 0, prefetch A regs for tile 1
    loadA(ar, 0);
    convstoreA(ar, 0);
    stageB(0, 0);
    loadA(ar, 1);

    const int NSTEP = Dd / TK;     // 16
    int cur = 0;
    for (int t = 0; t < NSTEP; ++t) {
#if USE_ASYNC
        __builtin_amdgcn_s_wait_asynccnt(0);
#endif
        __syncthreads();
        int nxt = cur ^ 1;
        if (t + 1 < NSTEP) {
            stageB(t + 1, nxt);          // async, flies under the WMMAs below
            convstoreA(ar, nxt);         // regs loaded last iteration
            if (t + 2 < NSTEP) loadA(ar, t + 2);
        }

        const unsigned short* Ah = smem + cur * BUFS;
        const unsigned short* Al = Ah + ABUF;
        const unsigned short* Bh = Ah + 2 * ABUF;
        const unsigned short* Bl = Bh + BBUF;

        // Fragments (ISA 16-bit layouts), pure ds_load_b128:
        Frag fah[2], fal[2], fbh[2], fbl[2];
        #pragma unroll
        for (int mi = 0; mi < 2; ++mi) {
            int base = (m0 + mi * 16 + lh) * SA + sel * 8;
            fah[mi].q[0] = *(const v4u*)(Ah + base);
            fah[mi].q[1] = *(const v4u*)(Ah + base + 16);
            fal[mi].q[0] = *(const v4u*)(Al + base);
            fal[mi].q[1] = *(const v4u*)(Al + base + 16);
        }
        #pragma unroll
        for (int ni = 0; ni < 2; ++ni) {
            int base = (n0 + ni * 16 + lh) * SA + sel * 8;
            fbh[ni].q[0] = *(const v4u*)(Bh + base);
            fbh[ni].q[1] = *(const v4u*)(Bh + base + 16);
            fbl[ni].q[0] = *(const v4u*)(Bl + base);
            fbl[ni].q[1] = *(const v4u*)(Bl + base + 16);
        }

        // bf16x3: hi*hi + lo*hi + hi*lo
        #pragma unroll
        for (int mi = 0; mi < 2; ++mi)
            #pragma unroll
            for (int ni = 0; ni < 2; ++ni) {
                acc[mi][ni] = __builtin_amdgcn_wmma_f32_16x16x32_bf16(
                    false, fah[mi].v, false, fbh[ni].v, (short)0, acc[mi][ni], false, false);
                acc[mi][ni] = __builtin_amdgcn_wmma_f32_16x16x32_bf16(
                    false, fal[mi].v, false, fbh[ni].v, (short)0, acc[mi][ni], false, false);
                acc[mi][ni] = __builtin_amdgcn_wmma_f32_16x16x32_bf16(
                    false, fah[mi].v, false, fbl[ni].v, (short)0, acc[mi][ni], false, false);
            }
        cur = nxt;
    }

    // Epilogue: lane half sel -> M = r + 8*sel, N = lh.
    #pragma unroll
    for (int mi = 0; mi < 2; ++mi) {
        #pragma unroll
        for (int r = 0; r < 8; ++r) {
            int Mg = mbase + m0 + mi * 16 + sel * 8 + r;
            float xx = x2[(size_t)h * BN + Mg];
            float best = -3.4e38f;
            int bestn = 0x7fffffff;
            #pragma unroll
            for (int ni = 0; ni < 2; ++ni) {
                int Ng = nbase + n0 + ni * 16 + lh;
                float yy = y2[h * Cc + Ng];
                float s = xx + yy - 2.f * acc[mi][ni][r];
                float dv = -sqrtf(fmaxf(s, 0.f));
                __builtin_nontemporal_store(
                    dv, dist_out + ((size_t)h * BN + Mg) * Cc + Ng);
                if (dv > best || (dv == best && Ng < bestn)) { best = dv; bestn = Ng; }
            }
            #pragma unroll
            for (int off = 8; off; off >>= 1) {
                float ov = __shfl_xor(best, off, 32);
                int   on = __shfl_xor(bestn, off, 32);
                if (ov > best || (ov == best && on < bestn)) { best = ov; bestn = on; }
            }
            if (lh == 0) {
                int pt = (nbase + n0) >> 5;
                size_t pi = ((size_t)h * BN + Mg) * NPART + pt;
                pmax[pi] = best;
                pidx[pi] = bestn;
            }
        }
    }
}

// ---------------------------------------------------------------------------
// Kernel 3: reduce 64 partials/row -> embed_ind (as float) + gather quantize.
// ---------------------------------------------------------------------------
__global__ __launch_bounds__(256) void reduce_gather_kernel(
    const float* __restrict__ e,
    const float* __restrict__ pmax, const int* __restrict__ pidx,
    float* __restrict__ outq, float* __restrict__ outind)
{
    int wave = threadIdx.x >> 5;
    int lane = threadIdx.x & 31;
    int row  = blockIdx.x * 8 + wave;
    if (row >= Hh * BN) return;
    int h = row / BN;

    size_t base = (size_t)row * NPART;
    float best = -3.4e38f;
    int bestn = 0x7fffffff;
    #pragma unroll
    for (int k = 0; k < 2; ++k) {
        int p = lane + k * 32;
        float v = pmax[base + p];
        int   n = pidx[base + p];
        if (v > best || (v == best && n < bestn)) { best = v; bestn = n; }
    }
    #pragma unroll
    for (int off = 16; off; off >>= 1) {
        float ov = __shfl_xor(best, off, 32);
        int   on = __shfl_xor(bestn, off, 32);
        if (ov > best || (ov == best && on < bestn)) { best = ov; bestn = on; }
    }
    bestn = __shfl(bestn, 0, 32);
    if (lane == 0) outind[row] = (float)bestn;

    const float4* src = (const float4*)(e + ((size_t)h * Cc + bestn) * Dd);
    float4*       dst = (float4*)(outq + (size_t)row * Dd);
    #pragma unroll
    for (int j = 0; j < 4; ++j) dst[lane + j * 32] = src[lane + j * 32];
}

// ---------------------------------------------------------------------------
extern "C" void kernel_launch(void* const* d_in, const int* in_sizes, int n_in,
                              void* d_out, int out_size, void* d_ws, size_t ws_size,
                              hipStream_t stream)
{
    const float* x = (const float*)d_in[0];   // (H, B, N, D) fp32
    const float* e = (const float*)d_in[1];   // (H, C, D)    fp32
    float* out = (float*)d_out;
    float* ws  = (float*)d_ws;

    float* x2 = ws + X2_OFF;
    float* y2 = ws + Y2_OFF;
    float* pm = ws + PM_OFF;
    int*   pi = (int*)(ws + PI_OFF);
    unsigned short* eh_plane = (unsigned short*)(ws + EH_OFF);
    unsigned short* el_plane = eh_plane + EPLANE;

    // 1) norms + embed bf16 split: 81920 rows, 8 per block
    int nrows = Hh * BN + Hh * Cc;
    prep_kernel<<<nrows / 8, 256, 0, stream>>>(x, e, ws, eh_plane, el_plane);

    // 2) GEMM + dist + partial argmax
    dim3 grid(BN / TM, Cc / TN, Hh);   // 64 x 32 x 8
    gemm_dist_kernel<<<grid, 256, 0, stream>>>(
        x, eh_plane, el_plane, x2, y2, out + DIST_OFF, pm, pi);

    // 3) final argmax + gather
    reduce_gather_kernel<<<(Hh * BN) / 8, 256, 0, stream>>>(
        e, pm, pi, out + Q_OFF, out + IND_OFF);
}
